// Recurrence_51067161150169
// MI455X (gfx1250) — matrix operand order repaired and checked by reading, not compiled, not run on hardware
//
#include <hip/hip_runtime.h>

// ---------------------------------------------------------------------------
// Problem constants (reference: B=2, S=4096, D=2048)
// ---------------------------------------------------------------------------
#define B_DIM 2
#define S_DIM 4096
#define D_DIM 2048
#define M_TOT (B_DIM * S_DIM)   // 8192 rows

// GEMM tiling: block = 128(M) x 32(N), K-step 32, 256 threads = 8 waves.
// Each wave owns 16 M-rows (A direct from global) and both 16-col N-subtiles.
#define BM 128
#define BN 32
#define BK 32
#define THREADS 256
#define KSTEPS (D_DIM / BK)     // 64

typedef __attribute__((ext_vector_type(16))) __bf16 v16bf;
typedef __attribute__((ext_vector_type(8)))  float  v8f;

union Frag16 { v16bf v; unsigned u[8]; uint4 q[2]; };

__device__ __forceinline__ unsigned short f2bf(float f) {
    unsigned u = __float_as_uint(f);
    u += 0x7FFFu + ((u >> 16) & 1u);      // round-to-nearest-even
    return (unsigned short)(u >> 16);
}
__device__ __forceinline__ float bf2f(unsigned short h) {
    return __uint_as_float(((unsigned)h) << 16);
}
__device__ __forceinline__ float sigmoidf_(float x) {
    return 1.0f / (1.0f + __expf(-x));
}

// CDNA5 async global->LDS copy, 16B per lane (tracked by ASYNCcnt).
__device__ __forceinline__ void async_copy_b128(unsigned lds_off, const void* gsrc) {
    asm volatile("global_load_async_to_lds_b128 %0, %1, off"
                 :: "v"(lds_off), "v"(gsrc)
                 : "memory");
}
__device__ __forceinline__ void wait_async0() {
    asm volatile("s_wait_asynccnt 0x0" ::: "memory");
}

// A fragment (16x32 bf16, ISA 7.12.2 layout) loaded DIRECTLY from global:
// lane L (L<16): M=L, K = 0..7 & 16..23 ; lane L+16: M=L, K = 8..15 & 24..31.
// Two contiguous 16B loads per lane.
__device__ __forceinline__ Frag16 load_a_frag_global(const unsigned short* Xb,
                                                     int mrow, int k0, int lane) {
    Frag16 a;
    const int kh = (lane >> 4) << 3;   // 0 or 8
    const size_t base = (size_t)mrow * D_DIM + k0 + kh;
    a.q[0] = *reinterpret_cast<const uint4*>(&Xb[base]);        // K kh..kh+7
    a.q[1] = *reinterpret_cast<const uint4*>(&Xb[base + 16]);   // K 16+kh..
    return a;
}

// B fragment (32x16 KxN) from LDS tile staged as Wt rows: Bs[n][k], stride BK.
// lanes 0-15: N=lane, K 0..7 & 16..23 ; lanes 16-31: N=lane-16, K 8..15 & 24..31.
__device__ __forceinline__ Frag16 load_b_frag(const unsigned short* Bs,
                                              int nbase, int lane) {
    Frag16 b;
    const int ncol = nbase + (lane & 15);
    const int kh = (lane >> 4) << 3;   // 0 or 8
    const uint4* bp = reinterpret_cast<const uint4*>(&Bs[ncol * BK + kh]);
    b.q[0] = bp[0];
    b.q[1] = bp[2];
    return b;
}

// ---------------------------------------------------------------------------
// One-time conversion kernels (hoist f32->bf16 out of the GEMM hot loops)
// ---------------------------------------------------------------------------
// Elementwise f32 -> bf16 (x activations). One float4 per thread.
__global__ void __launch_bounds__(256)
cvt_bf16_kernel(const float* __restrict__ in, unsigned short* __restrict__ out) {
    size_t i = ((size_t)blockIdx.x * blockDim.x + threadIdx.x) * 4;
    float4 v = *reinterpret_cast<const float4*>(&in[i]);
    uint2 pk;
    pk.x = (unsigned)f2bf(v.x) | ((unsigned)f2bf(v.y) << 16);
    pk.y = (unsigned)f2bf(v.z) | ((unsigned)f2bf(v.w) << 16);
    *reinterpret_cast<uint2*>(&out[i]) = pk;
}

// Transpose + convert: out[n][k] = bf16(in[k][n]), via padded LDS tile.
__global__ void __launch_bounds__(256)
transpose_bf16_kernel(const float* __restrict__ in, unsigned short* __restrict__ out) {
    __shared__ unsigned short tile[32][33];
    const int k0 = blockIdx.x * 32;
    const int n0 = blockIdx.y * 32;
    const int row = threadIdx.x >> 3;          // 0..31
    const int col = (threadIdx.x & 7) << 2;    // 0,4,..,28

    float4 v = *reinterpret_cast<const float4*>(&in[(size_t)(k0 + row) * D_DIM + n0 + col]);
    tile[row][col + 0] = f2bf(v.x);
    tile[row][col + 1] = f2bf(v.y);
    tile[row][col + 2] = f2bf(v.z);
    tile[row][col + 3] = f2bf(v.w);
    __syncthreads();

    unsigned short a0 = tile[col + 0][row];
    unsigned short a1 = tile[col + 1][row];
    unsigned short a2 = tile[col + 2][row];
    unsigned short a3 = tile[col + 3][row];
    uint2 pk;
    pk.x = (unsigned)a0 | ((unsigned)a1 << 16);
    pk.y = (unsigned)a2 | ((unsigned)a3 << 16);
    *reinterpret_cast<uint2*>(&out[(size_t)(n0 + row) * D_DIM + k0 + col]) = pk;
}

// ---------------------------------------------------------------------------
// Kernel 1: fused 3-way gate GEMM (bf16 WMMA, f32 acc), async double-buffered
// B staging.  r=silu(xWr)->bf16, f=sigmoid(xWf)->f32, c=silu(xWc)->f32.
// ---------------------------------------------------------------------------
__global__ void __launch_bounds__(THREADS)
gate_gemm_kernel(const unsigned short* __restrict__ Xb,
                 const unsigned short* __restrict__ Wtr,   // [N][K] bf16
                 const unsigned short* __restrict__ Wtf,
                 const unsigned short* __restrict__ Wtc,
                 unsigned short* __restrict__ Rb,
                 float* __restrict__ Fo,
                 float* __restrict__ Co) {
    __shared__ unsigned short Bs[2][3][BN * BK];   // double-buffered, 12 KB

    const int tid  = threadIdx.x;
    const int lane = tid & 31;
    const int wave = tid >> 5;
    const int m0 = blockIdx.y * BM;
    const int n0 = blockIdx.x * BN;
    const int mrow = m0 + (wave << 4) + (lane & 15);

    const unsigned short* Wt[3] = { Wtr, Wtf, Wtc };

    // async-stage one 3-gate B tile (3 x 32n x 32k bf16 = 384 x 16B chunks)
    auto stage_b = [&](int buf, int k0) {
        #pragma unroll
        for (int c = tid; c < 384; c += THREADS) {
            int g   = c >> 7;            // gate
            int l   = c & 127;
            int row = l >> 2;            // n offset 0..31
            int col = (l & 3) << 3;      // k offset 0,8,16,24
            unsigned dst = (unsigned)(size_t)&Bs[buf][g][row * BK + col];
            async_copy_b128(dst, &Wt[g][(size_t)(n0 + row) * D_DIM + k0 + col]);
        }
    };

    const v8f vzero = {0.f,0.f,0.f,0.f,0.f,0.f,0.f,0.f};
    v8f acc[3][2];
    #pragma unroll
    for (int g = 0; g < 3; ++g)
        #pragma unroll
        for (int t = 0; t < 2; ++t) acc[g][t] = vzero;

    // prologue: stage tile 0
    stage_b(0, 0);
    wait_async0();
    __syncthreads();

    for (int ks = 0; ks < KSTEPS; ++ks) {
        const int cur = ks & 1;
        const int k0 = ks * BK;

        if (ks + 1 < KSTEPS) {
            stage_b(cur ^ 1, k0 + BK);   // async copies overlap the WMMAs below
            __builtin_prefetch(&Xb[(size_t)mrow * D_DIM + k0 + 2 * BK], 0, 0);
        }

        Frag16 a = load_a_frag_global(Xb, mrow, k0, lane);
        #pragma unroll
        for (int g = 0; g < 3; ++g) {
            #pragma unroll
            for (int t = 0; t < 2; ++t) {
                Frag16 b = load_b_frag(Bs[cur][g], t << 4, lane);
                acc[g][t] = __builtin_amdgcn_wmma_f32_16x16x32_bf16(
                    false, a.v, false, b.v, (short)0, acc[g][t], false, false);
            }
        }

        wait_async0();       // next buffer resident
        __syncthreads();     // all waves done reading cur
    }

    // epilogue (C layout: VGPR i, lanes 0-15 -> M=i, lanes 16-31 -> M=8+i)
    const int rbase = m0 + (wave << 4) + ((lane >> 4) << 3);
    const int cbase = n0 + (lane & 15);
    #pragma unroll
    for (int t = 0; t < 2; ++t) {
        int col = cbase + (t << 4);
        #pragma unroll
        for (int i = 0; i < 8; ++i) {
            size_t idx = (size_t)(rbase + i) * D_DIM + col;
            float vr = acc[0][t][i];
            float vf = acc[1][t][i];
            float vc = acc[2][t][i];
            Rb[idx] = f2bf(vr * sigmoidf_(vr));   // silu -> bf16
            Fo[idx] = sigmoidf_(vf);
            Co[idx] = vc * sigmoidf_(vc);
        }
    }
}

// ---------------------------------------------------------------------------
// Kernel 2: gated linear recurrence along S, one thread per (b,d) channel.
// h_0 = c_0 ; h_t = f_t*h_{t-1} + (1-f_t)*c_t   (overwrites C with h)
// ---------------------------------------------------------------------------
__global__ void scan_kernel(const float* __restrict__ F, float* __restrict__ C) {
    int ch = blockIdx.x * blockDim.x + threadIdx.x;   // 0 .. B*D-1
    int b = ch / D_DIM;
    int d = ch - b * D_DIM;
    size_t base = (size_t)b * S_DIM * D_DIM + d;
    float h = C[base];
    for (int t = 1; t < S_DIM; ++t) {
        size_t idx = base + (size_t)t * D_DIM;
        float f = F[idx];
        float c = C[idx];
        h = f * h + (1.0f - f) * c;
        C[idx] = h;
    }
}

// ---------------------------------------------------------------------------
// Kernel 3: RMSNorm over D + gate with r, emit bf16 activations for last GEMM
// ---------------------------------------------------------------------------
__global__ void __launch_bounds__(256)
rmsnorm_gate_kernel(const float* __restrict__ H,
                    const unsigned short* __restrict__ Rb,
                    const float* __restrict__ ln_scale,
                    unsigned short* __restrict__ Gb) {
    __shared__ float red[256];
    const int row = blockIdx.x;
    const int tid = threadIdx.x;
    const size_t base = (size_t)row * D_DIM;

    float s = 0.f;
    #pragma unroll
    for (int j = 0; j < D_DIM / 256; ++j) {
        float v = H[base + tid + j * 256];
        s += v * v;
    }
    red[tid] = s;
    __syncthreads();
    #pragma unroll
    for (int st = 128; st > 0; st >>= 1) {
        if (tid < st) red[tid] += red[tid + st];
        __syncthreads();
    }
    const float inv = rsqrtf(red[0] * (1.0f / D_DIM) + 1e-5f);
    #pragma unroll
    for (int j = 0; j < D_DIM / 256; ++j) {
        int d = tid + j * 256;
        size_t idx = base + d;
        float g = H[idx] * inv * ln_scale[d];
        Gb[idx] = f2bf(bf2f(Rb[idx]) * g);
    }
}

// ---------------------------------------------------------------------------
// Kernel 4: output GEMM, bf16 activations @ Wo^T -> f32 d_out.  Same async
// double-buffered B staging, single weight matrix.
// ---------------------------------------------------------------------------
__global__ void __launch_bounds__(THREADS)
out_gemm_kernel(const unsigned short* __restrict__ Gb,
                const unsigned short* __restrict__ Wto,   // [N][K] bf16
                float* __restrict__ Y) {
    __shared__ unsigned short Bs[2][BN * BK];   // 4 KB

    const int tid  = threadIdx.x;
    const int lane = tid & 31;
    const int wave = tid >> 5;
    const int m0 = blockIdx.y * BM;
    const int n0 = blockIdx.x * BN;
    const int mrow = m0 + (wave << 4) + (lane & 15);

    auto stage_b = [&](int buf, int k0) {
        if (tid < 128) {
            int row = tid >> 2;          // n offset 0..31
            int col = (tid & 3) << 3;    // k offset
            unsigned dst = (unsigned)(size_t)&Bs[buf][row * BK + col];
            async_copy_b128(dst, &Wto[(size_t)(n0 + row) * D_DIM + k0 + col]);
        }
    };

    const v8f vzero = {0.f,0.f,0.f,0.f,0.f,0.f,0.f,0.f};
    v8f acc[2] = { vzero, vzero };

    stage_b(0, 0);
    wait_async0();
    __syncthreads();

    for (int ks = 0; ks < KSTEPS; ++ks) {
        const int cur = ks & 1;
        const int k0 = ks * BK;

        if (ks + 1 < KSTEPS) stage_b(cur ^ 1, k0 + BK);

        Frag16 a = load_a_frag_global(Gb, mrow, k0, lane);
        #pragma unroll
        for (int t = 0; t < 2; ++t) {
            Frag16 b = load_b_frag(Bs[cur], t << 4, lane);
            acc[t] = __builtin_amdgcn_wmma_f32_16x16x32_bf16(
                false, a.v, false, b.v, (short)0, acc[t], false, false);
        }

        wait_async0();
        __syncthreads();
    }

    const int rbase = m0 + (wave << 4) + ((lane >> 4) << 3);
    const int cbase = n0 + (lane & 15);
    #pragma unroll
    for (int t = 0; t < 2; ++t) {
        int col = cbase + (t << 4);
        #pragma unroll
        for (int i = 0; i < 8; ++i) {
            Y[(size_t)(rbase + i) * D_DIM + col] = acc[t][i];
        }
    }
}

// ---------------------------------------------------------------------------
// Host launcher
// ---------------------------------------------------------------------------
extern "C" void kernel_launch(void* const* d_in, const int* in_sizes, int n_in,
                              void* d_out, int out_size, void* d_ws, size_t ws_size,
                              hipStream_t stream) {
    const float* x  = (const float*)d_in[0];
    const float* Wr = (const float*)d_in[1];
    const float* Wf = (const float*)d_in[2];
    const float* Wc = (const float*)d_in[3];
    const float* Wo = (const float*)d_in[4];
    const float* ln = (const float*)d_in[5];
    float* y = (float*)d_out;

    // Workspace layout
    char* ws = (char*)d_ws;
    const size_t nMD = (size_t)M_TOT * D_DIM;      // 16.7M
    const size_t nDD = (size_t)D_DIM * D_DIM;      // 4.2M
    unsigned short* Xb  = (unsigned short*)ws;  ws += nMD * 2;
    unsigned short* Wtr = (unsigned short*)ws;  ws += nDD * 2;
    unsigned short* Wtf = (unsigned short*)ws;  ws += nDD * 2;
    unsigned short* Wtc = (unsigned short*)ws;  ws += nDD * 2;
    unsigned short* Wto = (unsigned short*)ws;  ws += nDD * 2;
    unsigned short* Rb  = (unsigned short*)ws;  ws += nMD * 2;
    unsigned short* Gb  = (unsigned short*)ws;  ws += nMD * 2;
    float* F = (float*)ws;                      ws += nMD * 4;
    float* C = (float*)ws;                      ws += nMD * 4;
    (void)ws_size; (void)in_sizes; (void)n_in; (void)out_size;

    // one-time precision/layout conversion (L2-resident afterwards)
    cvt_bf16_kernel<<<(unsigned)(nMD / 4 / 256), 256, 0, stream>>>(x, Xb);
    dim3 tgrid(D_DIM / 32, D_DIM / 32);
    transpose_bf16_kernel<<<tgrid, 256, 0, stream>>>(Wr, Wtr);
    transpose_bf16_kernel<<<tgrid, 256, 0, stream>>>(Wf, Wtf);
    transpose_bf16_kernel<<<tgrid, 256, 0, stream>>>(Wc, Wtc);
    transpose_bf16_kernel<<<tgrid, 256, 0, stream>>>(Wo, Wto);

    dim3 grid(D_DIM / BN, M_TOT / BM);   // 64 x 64 blocks
    gate_gemm_kernel<<<grid, THREADS, 0, stream>>>(Xb, Wtr, Wtf, Wtc, Rb, F, C);
    scan_kernel<<<(B_DIM * D_DIM) / 256, 256, 0, stream>>>(F, C);
    rmsnorm_gate_kernel<<<M_TOT, 256, 0, stream>>>(C, Rb, ln, Gb);
    out_gemm_kernel<<<grid, THREADS, 0, stream>>>(Gb, Wto, y);
}
